// GCN_21861383537347
// MI455X (gfx1250) — matrix operand order repaired and checked by reading, not compiled, not run on hardware
//
#include <hip/hip_runtime.h>
#include <hip/hip_bf16.h>

#define HID  128
#define NCLS 16
#define NGR  64
#define EPS_LN 1e-5f
#define ROWTILES 5   // 6250 row tiles / 5 = 1250 blocks exactly

typedef __attribute__((ext_vector_type(16))) __bf16 v16bf;
typedef __attribute__((ext_vector_type(8)))  float  v8f;

union Frag { uint4 q[2]; v16bf v; };   // 16 bf16 = 32 bytes

__device__ __forceinline__ unsigned short f2bf_u(float f) {
    unsigned u = __float_as_uint(f);
    return (unsigned short)((u + 0x7FFFu + ((u >> 16) & 1u)) >> 16);  // RNE
}

// ---------------------------------------------------------------- init
__global__ void gcn_init_k(float* deg, unsigned* gmaxkey, float* denom,
                           float* pooled, int N) {
    int i = blockIdx.x * blockDim.x + threadIdx.x;
    if (i < N) deg[i] = 1.0f;                       // self-loop
    if (i < NGR) { gmaxkey[i] = 0u; denom[i] = 0.f; }
    if (i < NGR * HID) pooled[i] = 0.f;
}

// ---------------------------------------------------------------- degree
__global__ void gcn_deg_k(const int* __restrict__ dst, float* deg, int E) {
    int e = blockIdx.x * blockDim.x + threadIdx.x;
    if (e < E) atomicAdd(&deg[dst[e]], 1.0f);
}

__global__ void gcn_dinv_k(const float* __restrict__ deg, float* dinv, int N) {
    int i = blockIdx.x * blockDim.x + threadIdx.x;
    if (i < N) dinv[i] = rsqrtf(deg[i]);
}

// ------------------------------------------------ fp32 -> bf16 (row-major)
__global__ void gcn_cvt_k(const float* __restrict__ in,
                          unsigned short* __restrict__ out, int n4) {
    int i = blockIdx.x * blockDim.x + threadIdx.x;   // float4 index
    if (i >= n4) return;
    float4 v = ((const float4*)in)[i];
    unsigned p0 = (unsigned)f2bf_u(v.x) | ((unsigned)f2bf_u(v.y) << 16);
    unsigned p1 = (unsigned)f2bf_u(v.z) | ((unsigned)f2bf_u(v.w) << 16);
    ((uint2*)out)[i] = make_uint2(p0, p1);
}

// ------------------------------------------------ weight transpose -> bf16
// W: [K][Ncols] fp32 row-major.  Wt: [Ncols][K] bf16 bits.
__global__ void gcn_wtrans_k(const float* __restrict__ W, unsigned short* Wt,
                             int K, int Ncols) {
    int n = blockIdx.x, k = threadIdx.x;
    if (k < K) Wt[(size_t)n * K + k] = f2bf_u(W[(size_t)k * Ncols + n]);
}

// ------------------------------------------------------------- WMMA GEMM
// out[nrows][NCOLS] = Ab[nrows][128](bf16) @ Wt^T (Wt: [NCOLS][128] bf16)
// Block: NCOLS/16 waves; wave w owns 16 columns; ROWTILES row tiles per
// block with the weight (B) fragments hoisted out of the row loop.
// NCOLS is compile-time so the 8 epilogue stores fold into immediate offsets.
template <int NCOLS>
__global__ void gcn_gemm_wmma_k(const unsigned short* __restrict__ Ab,
                                const unsigned short* __restrict__ Wt,
                                float* __restrict__ out, int nrows) {
    const int lane = threadIdx.x & 31;
    const int wave = threadIdx.x >> 5;
    const int half = lane >> 4;
    const int l16  = lane & 15;
    const int col  = wave * 16 + l16;
    const int tile0 = blockIdx.x * ROWTILES;

    // B fragments for all 4 K-steps (reused across row tiles)
    Frag b[4];
    const unsigned short* bcol = Wt + (size_t)col * HID;
#pragma unroll
    for (int kk = 0; kk < 4; ++kk) {
        const uint4* bp = (const uint4*)(bcol + kk * 32 + 8 * half);
        b[kk].q[0] = bp[0];
        b[kk].q[1] = bp[2];
    }

#pragma unroll
    for (int t = 0; t < ROWTILES; ++t) {
        const int row0 = (tile0 + t) * 16;
        if (row0 >= nrows) break;                       // block-uniform
        const unsigned short* arow = Ab + (size_t)(row0 + l16) * HID;
        v8f acc = {};
#pragma unroll
        for (int kk = 0; kk < 4; ++kk) {
            Frag a;   // elements 0-7: K=K0+8h.. ; 8-15: K=K0+16+8h..
            a.q[0] = *(const uint4*)(arow + kk * 32 + 8 * half);
            a.q[1] = *(const uint4*)(arow + kk * 32 + 16 + 8 * half);
            acc = __builtin_amdgcn_wmma_f32_16x16x32_bf16(
                      false, a.v, false, b[kk].v, (short)0, acc, false, false);
        }
        // C/D layout: VGPR r -> M=r (lanes 0-15), M=r+8 (lanes 16-31)
        float* obase = out + (size_t)(row0 + 8 * half) * NCOLS + col;
#pragma unroll
        for (int r = 0; r < 8; ++r)
            obase[r * NCOLS] = acc[r];    // constant-stride: immediate offsets
    }
}

// ------------------------------------------- self-loop term: B = A * dinv^2
__global__ void gcn_selfinit_k(const float* __restrict__ h,
                               const float* __restrict__ dinv,
                               float* __restrict__ out, int N) {
    int i = blockIdx.x * blockDim.x + threadIdx.x;     // float4 index
    if (i >= N * (HID / 4)) return;
    int n = i / (HID / 4);
    float d = dinv[n], w = d * d;
    float4 v = ((const float4*)h)[i];
    v.x *= w; v.y *= w; v.z *= w; v.w *= w;
    ((float4*)out)[i] = v;
}

// ------------------------------------- edge scatter: out[dst] += h[src]*norm
__global__ void gcn_edge_k(const int* __restrict__ src, const int* __restrict__ dst,
                           const float* __restrict__ dinv,
                           const float* __restrict__ h, float* out, int E) {
    int e = blockIdx.x * 8 + (threadIdx.x >> 5);
    if (e >= E) return;
    int lane = threadIdx.x & 31;
    int s = src[e], d = dst[e];
    float w = dinv[s] * dinv[d];
    float4 v = ((const float4*)(h + (size_t)s * HID))[lane];
    float* o = out + (size_t)d * HID + lane * 4;
    atomicAdd(o + 0, v.x * w); atomicAdd(o + 1, v.y * w);
    atomicAdd(o + 2, v.z * w); atomicAdd(o + 3, v.w * w);
}

// ------------------- LayerNorm(agg + bias) -> ReLU; fp32 + bf16 outputs
__global__ void gcn_ln_relu_k(const float* __restrict__ agg,
                              const float* __restrict__ bias,
                              const float* __restrict__ g,
                              const float* __restrict__ be,
                              float* __restrict__ out,
                              unsigned short* __restrict__ outb, int N) {
    int n = blockIdx.x * 8 + (threadIdx.x >> 5);
    if (n >= N) return;
    int lane = threadIdx.x & 31;
    float4 t  = ((const float4*)(agg + (size_t)n * HID))[lane];
    float4 bb = ((const float4*)bias)[lane];
    t.x += bb.x; t.y += bb.y; t.z += bb.z; t.w += bb.w;
    float s = t.x + t.y + t.z + t.w;
#pragma unroll
    for (int o = 16; o > 0; o >>= 1) s += __shfl_xor(s, o, 32);
    float mu = s * (1.0f / HID);
    float dx = t.x - mu, dy = t.y - mu, dz = t.z - mu, dw = t.w - mu;
    float ss = dx * dx + dy * dy + dz * dz + dw * dw;
#pragma unroll
    for (int o = 16; o > 0; o >>= 1) ss += __shfl_xor(ss, o, 32);
    float rstd = rsqrtf(ss * (1.0f / HID) + EPS_LN);
    float4 gg = ((const float4*)g)[lane];
    float4 ee = ((const float4*)be)[lane];
    float4 y;
    y.x = fmaxf(dx * rstd * gg.x + ee.x, 0.f);
    y.y = fmaxf(dy * rstd * gg.y + ee.y, 0.f);
    y.z = fmaxf(dz * rstd * gg.z + ee.z, 0.f);
    y.w = fmaxf(dw * rstd * gg.w + ee.w, 0.f);
    ((float4*)(out + (size_t)n * HID))[lane] = y;
    unsigned p0 = (unsigned)f2bf_u(y.x) | ((unsigned)f2bf_u(y.y) << 16);
    unsigned p1 = (unsigned)f2bf_u(y.z) | ((unsigned)f2bf_u(y.w) << 16);
    *(uint2*)(outb + (size_t)n * HID + lane * 4) = make_uint2(p0, p1);
}

// ------------- gate scalar: gate[n] = relu(g1+bg1) . Wg2 + bg2 ; seg-max key
__global__ void gcn_gate_k(const float* __restrict__ g1, const float* __restrict__ bg1,
                           const float* __restrict__ Wg2, const float* __restrict__ bg2,
                           const int* __restrict__ batch,
                           float* gate, unsigned* gmaxkey, int N) {
    int n = blockIdx.x * 8 + (threadIdx.x >> 5);
    if (n >= N) return;
    int lane = threadIdx.x & 31;
    const float* r = g1 + (size_t)n * (HID / 2);
    float p = fmaxf(r[lane]      + bg1[lane],      0.f) * Wg2[lane]
            + fmaxf(r[lane + 32] + bg1[lane + 32], 0.f) * Wg2[lane + 32];
#pragma unroll
    for (int o = 16; o > 0; o >>= 1) p += __shfl_xor(p, o, 32);
    if (lane == 0) {
        float gv = p + bg2[0];
        gate[n] = gv;
        unsigned u = __float_as_uint(gv);
        unsigned key = (u & 0x80000000u) ? ~u : (u | 0x80000000u);  // order-preserving
        atomicMax(&gmaxkey[batch[n]], key);
    }
}

__global__ void gcn_esum_k(const float* __restrict__ gate, const int* __restrict__ batch,
                           const unsigned* __restrict__ gmaxkey,
                           float* ebuf, float* denom, int N) {
    int n = blockIdx.x * blockDim.x + threadIdx.x;
    if (n >= N) return;
    int b = batch[n];
    unsigned key = gmaxkey[b];
    float m = (key & 0x80000000u) ? __uint_as_float(key & 0x7FFFFFFFu)
                                  : __uint_as_float(~key);
    float ev = expf(gate[n] - m);
    ebuf[n] = ev;
    atomicAdd(&denom[b], ev);
}

__global__ void gcn_pool_k(const float* __restrict__ h2, const float* __restrict__ ebuf,
                           const float* __restrict__ denom, const int* __restrict__ batch,
                           float* pooled, int N) {
    int n = blockIdx.x * 8 + (threadIdx.x >> 5);
    if (n >= N) return;
    int lane = threadIdx.x & 31;
    int b = batch[n];
    float a = ebuf[n] / denom[b];
    float4 v = ((const float4*)(h2 + (size_t)n * HID))[lane];
    float* o = pooled + (size_t)b * HID + lane * 4;
    atomicAdd(o + 0, v.x * a); atomicAdd(o + 1, v.y * a);
    atomicAdd(o + 2, v.z * a); atomicAdd(o + 3, v.w * a);
}

__global__ void gcn_cls_k(const float* __restrict__ pooled, const float* __restrict__ Wc,
                          const float* __restrict__ bc, float* __restrict__ out) {
    int i = blockIdx.x * blockDim.x + threadIdx.x;
    if (i >= NGR * NCLS) return;
    int b = i / NCLS, c = i % NCLS;
    const float* p = pooled + (size_t)b * HID;
    float s = bc[c];
#pragma unroll
    for (int k = 0; k < HID; ++k) s += p[k] * Wc[k * NCLS + c];
    out[i] = s;
}

// ==================================================================== launch
extern "C" void kernel_launch(void* const* d_in, const int* in_sizes, int n_in,
                              void* d_out, int out_size, void* d_ws, size_t ws_size,
                              hipStream_t stream) {
    const float* x     = (const float*)d_in[0];
    const int*   ei    = (const int*)d_in[1];
    const int*   batch = (const int*)d_in[2];
    const float* W1  = (const float*)d_in[3];
    const float* b1  = (const float*)d_in[4];
    const float* g1  = (const float*)d_in[5];
    const float* be1 = (const float*)d_in[6];
    const float* W2  = (const float*)d_in[7];
    const float* b2  = (const float*)d_in[8];
    const float* g2  = (const float*)d_in[9];
    const float* be2 = (const float*)d_in[10];
    const float* Wg1 = (const float*)d_in[11];
    const float* bg1 = (const float*)d_in[12];
    const float* Wg2 = (const float*)d_in[13];
    const float* bg2 = (const float*)d_in[14];
    const float* Wc  = (const float*)d_in[15];
    const float* bc  = (const float*)d_in[16];

    const int N = in_sizes[0] / HID;      // 100000 (multiple of 16)
    const int E = in_sizes[1] / 2;        // 1600000
    const int* src = ei;
    const int* dst = ei + E;

    // ---- workspace layout ----
    float* A = (float*)d_ws;                         // GEMM out / pre-agg
    float* B = A + (size_t)N * HID;                  // agg / gate1
    float* C = B + (size_t)N * HID;                  // LN fp32 out (h2)
    unsigned short* Ab = (unsigned short*)(C + (size_t)N * HID); // bf16 GEMM input
    float* deg  = (float*)(Ab + (size_t)N * HID);
    float* dinv = deg + N;
    float* gate = dinv + N;
    float* ebuf = gate + N;
    float* denom = ebuf + N;
    unsigned* gmaxkey = (unsigned*)(denom + NGR);
    float* pooled = (float*)(gmaxkey + NGR);
    unsigned short* Wt1  = (unsigned short*)(pooled + NGR * HID);
    unsigned short* Wt2  = Wt1 + HID * HID;
    unsigned short* Wgt1 = Wt2 + HID * HID;
    (void)ws_size; (void)n_in; (void)out_size;

    const int T = 256;
    const int gemmBlocks = (N / 16 + ROWTILES - 1) / ROWTILES;

    // init + degree normalization
    gcn_init_k<<<(N + T - 1) / T, T, 0, stream>>>(deg, gmaxkey, denom, pooled, N);
    gcn_deg_k<<<(E + T - 1) / T, T, 0, stream>>>(dst, deg, E);
    gcn_dinv_k<<<(N + T - 1) / T, T, 0, stream>>>(deg, dinv, N);
    // bf16 transposed weights + bf16 features
    gcn_wtrans_k<<<HID, HID, 0, stream>>>(W1, Wt1, HID, HID);
    gcn_wtrans_k<<<HID, HID, 0, stream>>>(W2, Wt2, HID, HID);
    gcn_wtrans_k<<<HID / 2, HID, 0, stream>>>(Wg1, Wgt1, HID, HID / 2);
    gcn_cvt_k<<<(N * (HID / 4) + T - 1) / T, T, 0, stream>>>(x, Ab, N * (HID / 4));

    // ---- layer 1 ----
    gcn_gemm_wmma_k<HID><<<gemmBlocks, 256, 0, stream>>>(Ab, Wt1, A, N);
    gcn_selfinit_k<<<(N * (HID / 4) + T - 1) / T, T, 0, stream>>>(A, dinv, B, N);
    gcn_edge_k<<<(E + 7) / 8, T, 0, stream>>>(src, dst, dinv, A, B, E);
    gcn_ln_relu_k<<<(N + 7) / 8, T, 0, stream>>>(B, b1, g1, be1, C, Ab, N);

    // ---- layer 2 ----
    gcn_gemm_wmma_k<HID><<<gemmBlocks, 256, 0, stream>>>(Ab, Wt2, A, N);
    gcn_selfinit_k<<<(N * (HID / 4) + T - 1) / T, T, 0, stream>>>(A, dinv, B, N);
    gcn_edge_k<<<(E + 7) / 8, T, 0, stream>>>(src, dst, dinv, A, B, E);
    gcn_ln_relu_k<<<(N + 7) / 8, T, 0, stream>>>(B, b2, g2, be2, C, Ab, N);  // C=h2

    // ---- attention pooling ----
    gcn_gemm_wmma_k<HID / 2><<<gemmBlocks, 128, 0, stream>>>(Ab, Wgt1, B, N);
    gcn_gate_k<<<(N + 7) / 8, T, 0, stream>>>(B, bg1, Wg2, bg2, batch, gate, gmaxkey, N);
    gcn_esum_k<<<(N + T - 1) / T, T, 0, stream>>>(gate, batch, gmaxkey, ebuf, denom, N);
    gcn_pool_k<<<(N + 7) / 8, T, 0, stream>>>(C, ebuf, denom, batch, pooled, N);

    // ---- classifier ----
    gcn_cls_k<<<(NGR * NCLS + T - 1) / T, T, 0, stream>>>(pooled, Wc, bc, (float*)d_out);
}
